// TwoLayerNet_73057393705666
// MI455X (gfx1250) — matrix operand order repaired and checked by reading, not compiled, not run on hardware
//
#include <hip/hip_runtime.h>

// ---------------------------------------------------------------------------
// RBF attention, fused flash-style, bf16 WMMA on gfx1250 (wave32).
// out = softmax(-(|x|^2 + |k|^2 - 2 x.k^T)/t) @ values
// N=32768, D=128, H=8192, C=100 (padded to 112 for 16-wide WMMA tiles).
//
// Prepass converts keys -> bf16 and values -> transposed/padded bf16 in
// workspace; the main loop stages chunks into LDS with
// global_load_async_to_lds_b128 (ASYNCcnt), keeping VALU/VGPRs free for the
// WMMA stream.
// ---------------------------------------------------------------------------

typedef __attribute__((ext_vector_type(16))) __bf16 v16bf;
typedef __attribute__((ext_vector_type(8)))  float  v8f;

union Frag {            // one WMMA bf16 A/B operand: 16 bf16 = 8 VGPRs
    v16bf v;
    __bf16 e[16];
    uint4  q[2];
};

#define N_ROWS 32768
#define D_DIM  128
#define H_KEYS 8192
#define C_COLS 100
#define C_PAD  112
#define BH     64          // keys per chunk
#define LOG2E  1.44269504088896340736f

// async global->LDS copy, 16B per lane, tracked by ASYNCcnt
__device__ __forceinline__ void async_b128(uint32_t lds_addr, const void* gptr) {
    asm volatile("global_load_async_to_lds_b128 %0, %1, off"
                 :: "v"(lds_addr), "v"(gptr)
                 : "memory");
}
__device__ __forceinline__ void wait_async0() {
    asm volatile("s_wait_asynccnt 0x0" ::: "memory");
}

// ---------------- prepass 1: squared norms of x rows and key rows ----------
__global__ void sq_norms_kernel(const float* __restrict__ x,
                                const float* __restrict__ keys,
                                float* __restrict__ xsq,
                                float* __restrict__ ksq) {
    const int wave = threadIdx.x >> 5;
    const int lane = threadIdx.x & 31;
    const int row  = blockIdx.x * 8 + wave;            // one wave per row
    const float* src;
    float* dst;
    if (row < N_ROWS) { src = x    + row * D_DIM;            dst = xsq + row; }
    else              { src = keys + (row - N_ROWS) * D_DIM; dst = ksq + (row - N_ROWS); }
    float s = 0.0f;
    #pragma unroll
    for (int d = lane; d < D_DIM; d += 32) { float v = src[d]; s += v * v; }
    #pragma unroll
    for (int m = 16; m >= 1; m >>= 1) s += __shfl_xor(s, m, 32);
    if (lane == 0) *dst = s;
}

// ---------------- prepass 2: one-time bf16 packing -------------------------
// keys_bf[h*D + d]            = bf16(keys[h][d])
// vt_bf  [c*H + h] (c<C_PAD)  = bf16(values[h][c]) or 0 for pad columns
__global__ void pack_bf16_kernel(const float* __restrict__ keys,
                                 const float* __restrict__ values,
                                 __bf16* __restrict__ keys_bf,
                                 __bf16* __restrict__ vt_bf) {
    const int i = blockIdx.x * 256 + threadIdx.x;
    if (i < H_KEYS * D_DIM) keys_bf[i] = (__bf16)keys[i];
    if (i < C_PAD * H_KEYS) {
        const int c = i / H_KEYS;
        const int h = i - c * H_KEYS;
        vt_bf[i] = (c < C_COLS) ? (__bf16)values[h * C_COLS + c] : (__bf16)0.0f;
    }
}

// ---------------- main fused kernel ----------------------------------------
__launch_bounds__(256, 1)
__global__ void rbf_attn_kernel(const float* __restrict__ x,
                                const __bf16* __restrict__ keys_bf,
                                const __bf16* __restrict__ vt_bf,
                                const float* __restrict__ temperature,
                                const float* __restrict__ xsq,
                                const float* __restrict__ ksq,
                                float* __restrict__ out) {
    __shared__ __align__(16) __bf16 keys_lds[BH * D_DIM];      // 16 KB
    __shared__ __align__(16) __bf16 vt_lds[C_PAD * BH];        // 14 KB (values^T)
    __shared__ __align__(16) __bf16 p_lds[8 * 16 * BH];        // 16 KB (per-wave P tiles)

    const int tid  = threadIdx.x;
    const int wave = tid >> 5;
    const int lane = tid & 31;
    const int l16  = lane & 15;
    const int hi   = lane >> 4;                 // lane half (VGPR-layout split)

    const int   rowBase = blockIdx.x * 128 + wave * 16;
    const float sscale  = (1.0f / temperature[0]) * LOG2E;   // work in log2 space

    // LDS byte addresses (low 32 bits of the generic shared pointer)
    const uint32_t keys_lds_a = (uint32_t)(uintptr_t)(void*)keys_lds;
    const uint32_t vt_lds_a   = (uint32_t)(uintptr_t)(void*)vt_lds;

    // Persistent bf16 A-fragments of my 16 x-rows (4 k-steps of 32).
    // 16-bit A 16x32 layout: lanes 0-15 hold K {ks*32+0..7, +16..23},
    // lanes 16-31 hold {+8..15, +24..31} (ISA 7.12.2).
    Frag afrag[4];
    {
        const float* xr = x + (rowBase + l16) * D_DIM;
        #pragma unroll
        for (int ks = 0; ks < 4; ++ks) {
            const int lo = ks * 32 + hi * 8;
            #pragma unroll
            for (int j = 0; j < 8; ++j) afrag[ks].e[j]     = (__bf16)xr[lo + j];
            #pragma unroll
            for (int j = 0; j < 8; ++j) afrag[ks].e[8 + j] = (__bf16)xr[lo + 16 + j];
        }
    }
    float xsq8[8];
    #pragma unroll
    for (int r = 0; r < 8; ++r) xsq8[r] = xsq[rowBase + r + hi * 8];

    float mrow[8], lrow[8];
    #pragma unroll
    for (int r = 0; r < 8; ++r) { mrow[r] = -3.0e38f; lrow[r] = 0.0f; }
    v8f oacc[7];
    #pragma unroll
    for (int c = 0; c < 7; ++c) oacc[c] = {};

    for (int h0 = 0; h0 < H_KEYS; h0 += BH) {
        __syncthreads();   // previous iteration's LDS readers are done

        // ---- async-stage keys chunk: 64x128 bf16 = 16 KB, contiguous ----
        {
            const char* kg = (const char*)(keys_bf + (size_t)h0 * D_DIM);
            #pragma unroll
            for (int j = 0; j < 4; ++j)
                async_b128(keys_lds_a + tid * 16 + j * 4096, kg + tid * 16 + j * 4096);
        }
        // ---- async-stage values^T chunk: 112 rows x 128 B = 14 KB ----
        if (tid < 224) {
            const int c    = tid >> 1;
            const int half = tid & 1;
            const char* vg = (const char*)(vt_bf + (size_t)c * H_KEYS + h0) + half * 64;
            const uint32_t vl = vt_lds_a + c * 128 + half * 64;
            #pragma unroll
            for (int j = 0; j < 4; ++j) async_b128(vl + j * 16, vg + j * 16);
        }
        // prefetch next chunk toward L2 while WMMAs run on this one
        if (h0 + BH < H_KEYS) {
            __builtin_prefetch(keys_bf + (size_t)(h0 + BH) * D_DIM + tid * 32, 0, 3);
        }
        wait_async0();       // my async writes to LDS are complete
        __syncthreads();     // everyone's are visible

        // ---- GEMM1: scores for 16 rows x 64 keys (4 col-tiles, 4 k-steps) ----
        v8f   sacc[4];
        float ksq4[4];
        #pragma unroll
        for (int ct = 0; ct < 4; ++ct) {
            v8f acc = {};
            #pragma unroll
            for (int ks = 0; ks < 4; ++ks) {
                // 16-bit B 32x16 layout: lanes 0-15 hold K {0..7, 16..23}, lanes 16-31 K {8..15, 24..31}
                Frag b;
                const uint4* bp =
                    (const uint4*)(keys_lds + (ct * 16 + l16) * D_DIM + ks * 32 + hi * 8);
                b.q[0] = bp[0]; b.q[1] = bp[2];
                acc = __builtin_amdgcn_wmma_f32_16x16x32_bf16(
                          false, afrag[ks].v, false, b.v, (short)0, acc, false, false);
            }
            sacc[ct] = acc;
            ksq4[ct] = ksq[h0 + ct * 16 + l16];
        }
        // logits (log2 space): -max(|x|^2+|k|^2-2xk, 0) * (log2e / t)
        #pragma unroll
        for (int ct = 0; ct < 4; ++ct) {
            #pragma unroll
            for (int r = 0; r < 8; ++r) {
                float sq = xsq8[r] + ksq4[ct] - 2.0f * sacc[ct][r];
                sq = fmaxf(sq, 0.0f);
                sacc[ct][r] = -sq * sscale;
            }
        }
        // ---- online softmax: row max, rescale, exp, row sum ----
        float mnew[8], alpha[8];
        #pragma unroll
        for (int r = 0; r < 8; ++r) {
            float t = fmaxf(fmaxf(sacc[0][r], sacc[1][r]), fmaxf(sacc[2][r], sacc[3][r]));
            #pragma unroll
            for (int m = 8; m >= 1; m >>= 1) t = fmaxf(t, __shfl_xor(t, m, 32));
            mnew[r]  = fmaxf(mrow[r], t);
            alpha[r] = __builtin_amdgcn_exp2f(mrow[r] - mnew[r]);
            mrow[r]  = mnew[r];
        }
        #pragma unroll
        for (int ct = 0; ct < 4; ++ct)
            #pragma unroll
            for (int r = 0; r < 8; ++r)
                sacc[ct][r] = __builtin_amdgcn_exp2f(sacc[ct][r] - mnew[r]);
        #pragma unroll
        for (int r = 0; r < 8; ++r) {
            float t = sacc[0][r] + sacc[1][r] + sacc[2][r] + sacc[3][r];
            #pragma unroll
            for (int m = 8; m >= 1; m >>= 1) t += __shfl_xor(t, m, 32);
            lrow[r] = lrow[r] * alpha[r] + t;
        }
        #pragma unroll
        for (int cc = 0; cc < 7; ++cc)
            #pragma unroll
            for (int r = 0; r < 8; ++r)
                oacc[cc][r] *= alpha[r];

        // ---- relayout P via per-wave LDS tile (D-layout -> A-layout) ----
        // Same-wave DS ops complete in order, so no workgroup barrier needed.
        __bf16* pw = p_lds + wave * 16 * BH;
        #pragma unroll
        for (int ct = 0; ct < 4; ++ct)
            #pragma unroll
            for (int r = 0; r < 8; ++r)
                pw[(r + hi * 8) * BH + ct * 16 + l16] = (__bf16)sacc[ct][r];

        // ---- GEMM2: O(16 x 112) += P(16 x 64) @ V(64 x 112) ----
        #pragma unroll
        for (int ks2 = 0; ks2 < 2; ++ks2) {
            Frag pa;
            {
                const __bf16* base = pw + l16 * BH + ks2 * 32 + hi * 8;
                pa.q[0] = *(const uint4*)(base);        // K .. K+7
                pa.q[1] = *(const uint4*)(base + 16);   // K+16 .. K+23
            }
            #pragma unroll
            for (int cc = 0; cc < 7; ++cc) {
                Frag bv;
                const uint4* vp =
                    (const uint4*)(vt_lds + (cc * 16 + l16) * BH + ks2 * 32 + hi * 8);
                bv.q[0] = vp[0]; bv.q[1] = vp[2];
                oacc[cc] = __builtin_amdgcn_wmma_f32_16x16x32_bf16(
                               false, pa.v, false, bv.v, (short)0, oacc[cc], false, false);
            }
        }
    }

    // ---- epilogue: normalize by softmax denominator, store (drop pad cols) ----
    float invl[8];
    #pragma unroll
    for (int r = 0; r < 8; ++r) invl[r] = 1.0f / lrow[r];
    #pragma unroll
    for (int cc = 0; cc < 7; ++cc) {
        const int c = cc * 16 + l16;
        if (c < C_COLS) {
            #pragma unroll
            for (int r = 0; r < 8; ++r) {
                const int row = rowBase + r + hi * 8;
                out[row * C_COLS + c] = oacc[cc][r] * invl[r];
            }
        }
    }
}

// ---------------------------------------------------------------------------
extern "C" void kernel_launch(void* const* d_in, const int* in_sizes, int n_in,
                              void* d_out, int out_size, void* d_ws, size_t ws_size,
                              hipStream_t stream) {
    const float* x      = (const float*)d_in[0];
    const float* keys   = (const float*)d_in[1];
    const float* values = (const float*)d_in[2];
    const float* temp   = (const float*)d_in[3];
    float* out = (float*)d_out;

    // workspace layout (16B-aligned sections): ~4.1 MB total
    float*  xsq     = (float*)d_ws;                          // N floats
    float*  ksq     = xsq + N_ROWS;                          // H floats
    __bf16* keys_bf = (__bf16*)(ksq + H_KEYS);               // H*D bf16 (2 MB)
    __bf16* vt_bf   = keys_bf + (size_t)H_KEYS * D_DIM;      // C_PAD*H bf16 (1.75 MB)

    sq_norms_kernel<<<(N_ROWS + H_KEYS) / 8, 256, 0, stream>>>(x, keys, xsq, ksq);
    pack_bf16_kernel<<<(H_KEYS * D_DIM) / 256, 256, 0, stream>>>(keys, values,
                                                                 keys_bf, vt_bf);
    rbf_attn_kernel<<<N_ROWS / 128, 256, 0, stream>>>(x, keys_bf, vt_bf, temp,
                                                      xsq, ksq, out);
}
